// GRUModel_46136538694210
// MI455X (gfx1250) — matrix operand order repaired and checked by reading, not compiled, not run on hardware
//
#include <hip/hip_runtime.h>
#include <hip/hip_bf16.h>

typedef __bf16 bf16;
typedef __attribute__((ext_vector_type(16))) bf16 v16bf;
typedef __attribute__((ext_vector_type(8)))  float v8f;

#define BB 64
#define TT 512
#define DD 128
#define HH 512
#define KTOT (DD + HH)          // 640: [x | h] concat reduction
#define NBLK 16                 // one full cluster: 16 workgroups
#define NTHR 1024               // 32 waves per workgroup -> 512 waves total
#define WPB  32
#define BH   (BB * HH)          // 32768 state elements

// ---- workspace layout (bytes) ----
#define WCAT_ELEMS (3u * HH * KTOT)           // 983040 bf16
#define WCAT_BYTES ((size_t)WCAT_ELEMS * 2)   // 1,966,080
#define H_OFF      (WCAT_BYTES)               // f32 [B,H]
#define HBF_OFF    (H_OFF   + (size_t)BH * 4)
#define ZP_OFF     (HBF_OFF + (size_t)BH * 2) // 2 x f32 [B,H] z partials
#define RP_OFF     (ZP_OFF  + (size_t)BH * 8) // 2 x f32 [B,H] r partials
#define GP_OFF     (RP_OFF  + (size_t)BH * 8) // 4 x f32 [B,H] g partials
#define YP_OFF     (GP_OFF  + (size_t)BH * 16) // f32 [B][8] y partials

struct GruParams {
  const float* x;
  const float* bz; const float* br; const float* bh;
  const float* Uz; const float* Ur; const float* Uh;
  const float* out_w; const float* out_b;
  const bf16* Wcat;
  float* h; bf16* hbf;
  float* zp; float* rp; float* gp;   // partial pre-activations (2/2/4 planes)
  float* ypart;                      // [B][8] per-wave y partial dots
  float* ys; float* hs; float* zs; float* rs; float* leaks; float* rdiags;
};

// ---------------- fast activations (v_exp_f32 + v_rcp_f32) ----------------
__device__ inline float fsigmoid(float x) {
  return __builtin_amdgcn_rcpf(1.0f + __expf(-x));
}
__device__ inline float ftanh(float x) {
  return 2.0f * __builtin_amdgcn_rcpf(1.0f + __expf(-2.0f * x)) - 1.0f;
}

// ---------------- prep: pack bf16 [Wg|Ug] rows, zero state ----------------
__global__ void gru_prep(const float* Wz, const float* Wr, const float* Wh,
                         const float* Uz, const float* Ur, const float* Uh,
                         bf16* Wcat, float* h, bf16* hbf) {
  int gid = blockIdx.x * blockDim.x + threadIdx.x;
  if (gid < (int)WCAT_ELEMS) {
    int g   = gid / (HH * KTOT);
    int rem = gid - g * (HH * KTOT);
    int n   = rem / KTOT;
    int k   = rem - n * KTOT;
    const float* W = (g == 0) ? Wz : (g == 1) ? Wr : Wh;
    const float* U = (g == 0) ? Uz : (g == 1) ? Ur : Uh;
    float v = (k < DD) ? W[n * DD + k] : U[n * HH + (k - DD)];
    Wcat[gid] = (bf16)v;
  }
  if (gid < BH) { h[gid] = 0.0f; hbf[gid] = (bf16)0.0f; }
}

// ---------------- cluster-wide barrier (hardware split barrier) -----------
__device__ inline void csync() {
  __threadfence();                      // publish our writes
  __syncthreads();                      // workgroup rendezvous first
  __builtin_amdgcn_s_cluster_barrier(); // s_barrier_signal -3 ; s_barrier_wait -3
  __threadfence();                      // acquire
}

// ---------------- K-sliced 16x16 tile GEMM, double-buffered fragments -----
// A (16-bit 16x32): lane=M%16, kg=lane/16; K=kg*16+0..15 (contiguous, matches B).
// B (16-bit 32x16): lanes0-15 K=0..15, lanes16-31 K=16..31 (2 bf16/VGPR),
//   so each lane's fragment is one contiguous 32B run of the Wcat row.
template <int KLEN>
__device__ inline v8f gate_gemm(const bf16* __restrict__ As,
                                const bf16* __restrict__ wrow0, int lane) {
  const int kh = lane >> 4;
  const bf16* arow = As + (lane & 15) * KLEN + kh * 16;
  const bf16* wrow = wrow0 + kh * 16;
  constexpr int NIT = KLEN / 32;
  v16bf a[2], b[2];
  auto loadA = [&](v16bf& d, int k0) {
#pragma unroll
    for (int j = 0; j < 16; ++j) d[j] = arow[k0 + j];
  };
  auto loadB = [&](v16bf& d, int k0) {
#pragma unroll
    for (int j = 0; j < 16; ++j) d[j] = wrow[k0 + j];
  };
  loadA(a[0], 0); loadB(b[0], 0);
  v8f c = {};
#pragma unroll
  for (int i = 0; i < NIT; ++i) {
    if (i + 1 < NIT) {                 // prefetch next fragments before WMMA
      loadA(a[(i + 1) & 1], (i + 1) * 32);
      loadB(b[(i + 1) & 1], (i + 1) * 32);
    }
    c = __builtin_amdgcn_wmma_f32_16x16x32_bf16(false, a[i & 1], false, b[i & 1],
                                                (short)0, c, false, false);
  }
  return c;
}

// ---------------- persistent cluster scan kernel ----------------
__attribute__((amdgpu_cluster_dims(NBLK, 1, 1)))
__global__ __launch_bounds__(NTHR, 1) void gru_persistent(GruParams p) {
  __shared__ bf16 As[16 * 320];        // 10 KB A-operand stage
  const int tid  = threadIdx.x;
  const int lane = tid & 31;
  const int wib  = tid >> 5;           // wave in block = n-tile id
  const int blk  = blockIdx.x;
  const int wid  = blk * WPB + wib;

  for (int t = 0; t < TT; ++t) {
    // ==== Phase A: z,r partial pre-acts, K=640 split in 2 halves of 320 ====
    {
      // deterministic, fixed-order combine of last step's y partials
      if (t > 0 && blk == 0 && tid < BB) {
        float s = p.out_b[0];
#pragma unroll
        for (int j = 0; j < 8; ++j) s += p.ypart[tid * 8 + j];
        p.ys[(size_t)tid * TT + (t - 1)] = s;
      }
      const int gate = blk >> 3;                  // 0=z, 1=r
      const int mt   = (blk >> 1) & 3;
      const int kh2  = blk & 1;
      const int klo  = kh2 * 320;
      for (int i = tid; i < 16 * 320; i += NTHR) {
        const int row = i / 320;
        const int kg  = klo + (i - row * 320);
        const int bgl = mt * 16 + row;
        bf16 v;
        if (kg < DD) v = (bf16)p.x[((size_t)bgl * TT + t) * DD + kg];
        else         v = p.hbf[bgl * HH + (kg - DD)];
        As[i] = v;
      }
      __syncthreads();
      const int n = wib * 16 + (lane & 15);
      const bf16* wrow0 =
          p.Wcat + (size_t)gate * HH * KTOT + (size_t)n * KTOT + klo;
      v8f c = gate_gemm<320>(As, wrow0, lane);
      float* dst = (gate == 0 ? p.zp : p.rp) + (size_t)kh2 * BH;
      const int mbase = mt * 16 + (lane >> 4) * 8;
#pragma unroll
      for (int i = 0; i < 8; ++i) dst[(mbase + i) * HH + n] = c[i];
    }
    csync();

    // ==== Phase B: g partials, K=640 split in 4 quarters of 160; the ====
    // ==== r*h A-operand is reconstructed inline from the r partials.  ====
    {
      const int mt  = blk >> 2;
      const int kq  = blk & 3;
      const int klo = kq * 160;
      for (int i = tid; i < 16 * 160; i += NTHR) {
        const int row = i / 160;
        const int kg  = klo + (i - row * 160);
        const int bgl = mt * 16 + row;
        bf16 v;
        if (kg < DD) {
          v = (bf16)p.x[((size_t)bgl * TT + t) * DD + kg];
        } else {
          const int nh = kg - DD;
          const int hi = bgl * HH + nh;
          float rr = fsigmoid(p.rp[hi] + p.rp[BH + hi] + p.br[nh]);
          v = (bf16)(rr * p.h[hi]);
        }
        As[i] = v;
      }
      __syncthreads();
      const int n = wib * 16 + (lane & 15);
      const bf16* wrow0 =
          p.Wcat + (size_t)2 * HH * KTOT + (size_t)n * KTOT + klo;
      v8f c = gate_gemm<160>(As, wrow0, lane);
      float* dst = p.gp + (size_t)kq * BH;
      const int mbase = mt * 16 + (lane >> 4) * 8;
#pragma unroll
      for (int i = 0; i < 8; ++i) dst[(mbase + i) * HH + n] = c[i];
    }
    csync();

    // ==== Phase C: combine partials, update state, write all outputs ====
    {
      const int e0 = wid * 64;                     // 512 waves x 64 = B*H
      const int b  = e0 >> 9;
      const int nb = e0 & (HH - 1);
      const size_t obase = (size_t)b * (TT * HH) + (size_t)t * HH;
      float ysum = 0.0f;
#pragma unroll
      for (int q = 0; q < 2; ++q) {
        const int n  = nb + q * 32 + lane;
        const int hi = b * HH + n;
        float hp = p.h[hi];
        float z  = fsigmoid(p.zp[hi] + p.zp[BH + hi] + p.bz[n]);
        float r  = fsigmoid(p.rp[hi] + p.rp[BH + hi] + p.br[n]);
        float g  = ftanh(p.gp[hi] + p.gp[BH + hi] + p.gp[2 * BH + hi] +
                         p.gp[3 * BH + hi] + p.bh[n]);
        float hn = (1.0f - z) * hp + z * g;
        p.zs[obase + n]    = z;
        p.rs[obase + n]    = r;
        p.leaks[obase + n] = 1.0f - z;
        p.hs[obase + n]    = hn;
        float uzd = p.Uz[n * (HH + 1)];
        float urd = p.Ur[n * (HH + 1)];
        float uhd = p.Uh[n * (HH + 1)];
        float zpr = z * (1.0f - z), rpr = r * (1.0f - r), gpr = 1.0f - g * g;
        p.rdiags[obase + n] =
            (g - hp) * zpr * uzd + z * gpr * uhd * (r + hp * rpr * urd);
        p.h[hi]   = hn;
        p.hbf[hi] = (bf16)hn;
        ysum += hn * p.out_w[n];
      }
#pragma unroll
      for (int off = 16; off > 0; off >>= 1)
        ysum += __shfl_xor(ysum, off, 32);
      if (lane == 0) p.ypart[b * 8 + (wid & 7)] = ysum;  // deterministic slot
    }
    csync();
  }

  // final y combine for t = TT-1 (fixed order)
  if (blk == 0 && tid < BB) {
    float s = p.out_b[0];
#pragma unroll
    for (int j = 0; j < 8; ++j) s += p.ypart[tid * 8 + j];
    p.ys[(size_t)tid * TT + (TT - 1)] = s;
  }
}

extern "C" void kernel_launch(void* const* d_in, const int* in_sizes, int n_in,
                              void* d_out, int out_size, void* d_ws, size_t ws_size,
                              hipStream_t stream) {
  (void)in_sizes; (void)n_in; (void)out_size; (void)ws_size;
  const float* x  = (const float*)d_in[0];
  const float* Wz = (const float*)d_in[1];
  const float* bz = (const float*)d_in[2];
  const float* Wr = (const float*)d_in[3];
  const float* br = (const float*)d_in[4];
  const float* Wh = (const float*)d_in[5];
  const float* bh = (const float*)d_in[6];
  const float* Uz = (const float*)d_in[7];
  const float* Ur = (const float*)d_in[8];
  const float* Uh = (const float*)d_in[9];
  const float* ow = (const float*)d_in[10];
  const float* ob = (const float*)d_in[11];

  char* ws = (char*)d_ws;
  bf16*  Wcat  = (bf16*)(ws);
  float* h     = (float*)(ws + H_OFF);
  bf16*  hbf   = (bf16*)(ws + HBF_OFF);
  float* zp    = (float*)(ws + ZP_OFF);
  float* rp    = (float*)(ws + RP_OFF);
  float* gp    = (float*)(ws + GP_OFF);
  float* ypart = (float*)(ws + YP_OFF);

  float* out = (float*)d_out;
  const size_t BTH = (size_t)BB * TT * HH;
  float* ys = out;
  float* hs = out + (size_t)BB * TT;
  float* zs = hs + BTH;
  float* rs = zs + BTH;
  float* lk = rs + BTH;
  float* rd = lk + BTH;

  gru_prep<<<(WCAT_ELEMS + 255) / 256, 256, 0, stream>>>(
      Wz, Wr, Wh, Uz, Ur, Uh, Wcat, h, hbf);

  GruParams p;
  p.x = x; p.bz = bz; p.br = br; p.bh = bh;
  p.Uz = Uz; p.Ur = Ur; p.Uh = Uh;
  p.out_w = ow; p.out_b = ob;
  p.Wcat = Wcat; p.h = h; p.hbf = hbf;
  p.zp = zp; p.rp = rp; p.gp = gp; p.ypart = ypart;
  p.ys = ys; p.hs = hs; p.zs = zs; p.rs = rs; p.leaks = lk; p.rdiags = rd;

  gru_persistent<<<NBLK, NTHR, 0, stream>>>(p);
}
